// Multi_head_att_58033598103591
// MI455X (gfx1250) — matrix operand order repaired and checked by reading, not compiled, not run on hardware
//
#include <hip/hip_runtime.h>
#include <hip/hip_bf16.h>

typedef __attribute__((ext_vector_type(16))) __bf16 v16bf;
typedef __attribute__((ext_vector_type(8)))  float  v8f;
typedef __attribute__((ext_vector_type(4)))  unsigned int v4u;
typedef __attribute__((ext_vector_type(8)))  int v8i;
typedef __attribute__((ext_vector_type(4)))  int v4i;
typedef unsigned short u16;

#define DIMD   1024
#define SEQ    1024
#define NB     4
#define NH     16
#define DP     64

// ---------- helpers ----------

// A-fragment (16-bit A 16x32): lane holds row m=lane&15; element chunks at
// k = kbase+8h..+7  and  k = kbase+16+8h..+7   (h = lane>>4)
static __device__ __forceinline__ v16bf ldA(const u16* p) {
  union { uint4 u[2]; v16bf v; } t;
  t.u[0] = *reinterpret_cast<const uint4*>(p);
  t.u[1] = *reinterpret_cast<const uint4*>(p + 16);
  return t.v;
}
// B-fragment (16-bit B 32x16): lane holds column n=lane&15; element chunks at
// k = kbase+8h..+7  and  k = kbase+16+8h..+7   (h = lane>>4)
static __device__ __forceinline__ v16bf ldB(const u16* p) {
  union { uint4 u[2]; v16bf v; } t;
  t.u[0] = *reinterpret_cast<const uint4*>(p);
  t.u[1] = *reinterpret_cast<const uint4*>(p + 16);
  return t.v;
}
static __device__ __forceinline__ v8f wmma_bf16(v16bf a, v16bf b, v8f c) {
  return __builtin_amdgcn_wmma_f32_16x16x32_bf16(false, a, false, b,
                                                 (short)0, c, false, false);
}
static __device__ __forceinline__ u16 f2bf(float x) {
  return __builtin_bit_cast(u16, (__bf16)x);
}
// flat->LDS offset: low 32 bits of a flat LDS address are the workgroup-
// relative LDS byte offset (aperture lives in addr[63:32] only).
static __device__ __forceinline__ unsigned lds_off(const void* p) {
  return (unsigned)(unsigned long long)p;
}
// Async copy of 16 bytes per lane from global to LDS (ASYNCcnt-tracked).
static __device__ __forceinline__ void async_b128(unsigned ldsaddr,
                                                  const void* gaddr) {
  asm volatile("global_load_async_to_lds_b128 %0, %1, off"
               :: "v"(ldsaddr), "v"((unsigned long long)gaddr) : "memory");
}
// TDM: stage a 2D bf16 tile (64 rows x 32 elements, row stride 1024 elems)
// from global into contiguous LDS.  D# packing per CDNA5 ISA:
//  g0: [1:0]=count=1 | lds_addr[63:32] | global_addr[120:64] | type=2[127:126]
//  g1: data_size=1(2B)[17:16]; tensor_dim0=32[79:48]; tensor_dim1=64[111:80];
//      tile_dim0=32[127:112]; tile_dim1=64[143:128]; dim0_stride=1024[207:160]
//  remaining groups zero (tile dims 2..4 unused -> 2D)
static __device__ __forceinline__ void tdm_stage(const u16* gaddr,
                                                 unsigned ldsaddr) {
  const unsigned long long ga = (unsigned long long)gaddr;
  v4u g0;
  g0[0] = 1u;                                   // count=1, user descriptor
  g0[1] = ldsaddr;                              // lds_addr (bytes)
  g0[2] = (unsigned)ga;                         // global_addr[31:0]
  g0[3] = (unsigned)(ga >> 32) | (2u << 30);    // global_addr[56:32] | type=2
  v8i g1;
  g1[0] = (int)(1u << 16);                      // data_size = 2 bytes
  g1[1] = (int)(32u << 16);                     // tensor_dim0 = 32 (lo16)
  g1[2] = (int)(64u << 16);                     // tensor_dim1 = 64 (lo16)
  g1[3] = (int)(32u << 16);                     // tile_dim0 = 32
  g1[4] = 64;                                   // tile_dim1 = 64
  g1[5] = 1024;                                 // tensor_dim0_stride = 1024
  g1[6] = 0;
  g1[7] = 0;
  v4i z4 = {0, 0, 0, 0};
  v8i z8 = {0, 0, 0, 0, 0, 0, 0, 0};
  __builtin_amdgcn_tensor_load_to_lds(g0, g1, z4, z4, z8, 0);
}
// DPP row-rotate (rows of 16 lanes = exactly the WMMA C-layout row groups).
template <int N>
static __device__ __forceinline__ float row_ror(float x) {
  int i = __builtin_bit_cast(int, x);
  i = __builtin_amdgcn_update_dpp(0, i, 0x120 + N, 0xf, 0xf, true);
  return __builtin_bit_cast(float, i);
}
// all-reduce across the 16 lanes of a DPP row (rotate butterfly)
static __device__ __forceinline__ float row_max16(float x) {
  x = fmaxf(x, row_ror<1>(x));
  x = fmaxf(x, row_ror<2>(x));
  x = fmaxf(x, row_ror<4>(x));
  x = fmaxf(x, row_ror<8>(x));
  return x;
}
static __device__ __forceinline__ float row_sum16(float x) {
  x += row_ror<1>(x);
  x += row_ror<2>(x);
  x += row_ror<4>(x);
  x += row_ror<8>(x);
  return x;
}

// ---------- weight pack: W[k][n] fp32 -> Wt[n][k] bf16 ----------
__global__ void pack_wt(const float* __restrict__ W, u16* __restrict__ Wt) {
  int idx = blockIdx.x * 256 + threadIdx.x;      // 0 .. D*D-1
  int k = idx >> 10, n = idx & (DIMD - 1);
  Wt[n * DIMD + k] = f2bf(W[idx]);
}

// ---------- GEMM: C[4096x1024] = (A @ W^T + bias) * out_scale -----------
// Block = 8 waves computing 128 rows x 64 cols; all waves share one B tile
// (64 cols x 32 k = 4KB), double-buffered in LDS.
//   A_IS_F32 (QKV projections): staged by the Tensor Data Mover
//     (tensor_load_to_lds, TENSORcnt) issued by wave 0.
//   !A_IS_F32 (output projection): staged with per-lane
//     global_load_async_to_lds_b128 (ASYNCcnt).
// out_mode 0: bf16 head-split [B,H,S,DP]   (Q, K)
// out_mode 1: bf16 depth-major [B,H,DP,S]  (V)
// out_mode 2: fp32 row-major   [B*S, D]    (final output)
template <bool A_IS_F32>
__global__ void gemm_kern(const float* __restrict__ A32,
                          const u16*  __restrict__ A16,
                          const u16*  __restrict__ Bt,
                          const float* __restrict__ bias,
                          u16* __restrict__ out16,
                          float* __restrict__ out32,
                          int out_mode, float out_scale) {
  __shared__ u16 Bsh[2][64 * 32];                // 2 x 4KB

  const int lane = threadIdx.x;
  const int nl   = lane & 15;
  const int hf   = lane >> 4;
  const int wave = threadIdx.y;
  const int tmb  = blockIdx.x & 31;              // 32 row-blocks of 128 rows
  const int tn   = blockIdx.x >> 5;              // 16 col tiles of 64
  const int tm   = tmb * 8 + wave;
  const int arow = tm * 16 + nl;
  const int nbase = tn * 64;

  // per-lane staging assignment (async path): 64 rows x 64B, 4 lanes/row
  const int tid  = wave * 32 + lane;
  const int srow = tid >> 2;
  const int sseg = (tid & 3) * 8;
  const u16* gB  = Bt + (size_t)(nbase + srow) * DIMD + sseg;
  const unsigned lds0 = lds_off(&Bsh[0][srow * 32 + sseg]);
  const unsigned lds1 = lds_off(&Bsh[1][srow * 32 + sseg]);
  // uniform staging bases (TDM path)
  const u16* gBbase = Bt + (size_t)nbase * DIMD;
  const unsigned ldsb0 = lds_off(&Bsh[0][0]);
  const unsigned ldsb1 = lds_off(&Bsh[1][0]);

  v8f acc[4];
  #pragma unroll
  for (int j = 0; j < 4; ++j) acc[j] = (v8f)(0.0f);

  // prologue: stage kk=0 into buffer 0
  if constexpr (A_IS_F32) {
    if (wave == 0) tdm_stage(gBbase, ldsb0);
  } else {
    async_b128(lds0, gB);
  }

  int cur = 0;
  for (int kk = 0; kk < DIMD; kk += 32) {
    const bool has_next = (kk + 32) < DIMD;
    if (kk > 0) __syncthreads();                 // WAR: readers of buf[cur^1] done
    if constexpr (A_IS_F32) {
      if (wave == 0) {
        if (has_next) {
          tdm_stage(gBbase + kk + 32, cur ? ldsb0 : ldsb1);
          __builtin_amdgcn_s_wait_tensorcnt(1);  // in-order: older one retired
        } else {
          __builtin_amdgcn_s_wait_tensorcnt(0);
        }
      }
    } else {
      if (has_next) {
        async_b128(cur ? lds0 : lds1, gB + kk + 32);
        asm volatile("s_wait_asynccnt 0x1" ::: "memory");
      } else {
        asm volatile("s_wait_asynccnt 0x0" ::: "memory");
      }
    }
    __syncthreads();                             // buf[cur] visible to all waves

    v16bf a;
    if constexpr (A_IS_F32) {
      const float* p = A32 + (size_t)arow * DIMD + kk + 8 * hf;
      float4 f0 = *reinterpret_cast<const float4*>(p);
      float4 f1 = *reinterpret_cast<const float4*>(p + 4);
      float4 f2 = *reinterpret_cast<const float4*>(p + 16);
      float4 f3 = *reinterpret_cast<const float4*>(p + 20);
      a[0]=(__bf16)f0.x; a[1]=(__bf16)f0.y; a[2]=(__bf16)f0.z; a[3]=(__bf16)f0.w;
      a[4]=(__bf16)f1.x; a[5]=(__bf16)f1.y; a[6]=(__bf16)f1.z; a[7]=(__bf16)f1.w;
      a[8]=(__bf16)f2.x; a[9]=(__bf16)f2.y; a[10]=(__bf16)f2.z; a[11]=(__bf16)f2.w;
      a[12]=(__bf16)f3.x; a[13]=(__bf16)f3.y; a[14]=(__bf16)f3.z; a[15]=(__bf16)f3.w;
    } else {
      a = ldA(A16 + (size_t)arow * DIMD + kk + 8 * hf);
    }
    #pragma unroll
    for (int j = 0; j < 4; ++j)
      acc[j] = wmma_bf16(a, ldB(&Bsh[cur][(16 * j + nl) * 32 + 8 * hf]), acc[j]);
    cur ^= 1;
  }

  // epilogue: C element r of acc[j] is (M = tm*16 + r + 8*hf, N = nbase+16j+nl)
  #pragma unroll
  for (int j = 0; j < 4; ++j) {
    const int ncol = nbase + 16 * j + nl;
    const float bv = bias[ncol];
    #pragma unroll
    for (int r = 0; r < 8; ++r) {
      const int mrow = tm * 16 + r + 8 * hf;
      const float val = (acc[j][r] + bv) * out_scale;
      if (out_mode == 2) {
        out32[(size_t)mrow * DIMD + ncol] = val;
      } else {
        const int b = mrow >> 10, s = mrow & (SEQ - 1);
        const int head = ncol >> 6, dp = ncol & (DP - 1);
        size_t idx;
        if (out_mode == 0)
          idx = (((size_t)(b * NH + head) * SEQ) + s) * DP + dp;
        else
          idx = (((size_t)(b * NH + head) * DP) + dp) * SEQ + s;
        out16[idx] = f2bf(val);
      }
    }
  }
}

// ---------- fused flash attention over one (b,h), 16 query rows / wave ----
// Q was pre-scaled by 0.125*log2(e) at projection time, so the logits are
// already base-2: softmax uses raw v_exp_f32 (2^x) directly.
__global__ void attn_kernel(const u16* __restrict__ Qh,
                            const u16* __restrict__ Kh,
                            const u16* __restrict__ Vt,
                            u16* __restrict__ ctx) {
  __shared__ u16 lds[4][16 * 32];

  const int lane = threadIdx.x;
  const int nl   = lane & 15;
  const int hf   = lane >> 4;
  const int wv   = threadIdx.y;
  const int bh   = blockIdx.x;            // b*16 + h
  const int b    = bh >> 4;
  const int hh   = bh & 15;
  const int qt   = (blockIdx.y * 4 + wv) * 16;

  const u16* qp = Qh + ((size_t)bh * SEQ + qt + nl) * DP;
  const v16bf qa0 = ldA(qp + 8 * hf);
  const v16bf qa1 = ldA(qp + 32 + 8 * hf);

  v8f o[4];
  #pragma unroll
  for (int d = 0; d < 4; ++d) o[d] = (v8f)(0.0f);
  float mrow[8], lrow[8];
  #pragma unroll
  for (int r = 0; r < 8; ++r) { mrow[r] = -3.0e38f; lrow[r] = 0.0f; }

  const u16* Kbh = Kh + (size_t)bh * SEQ * DP;
  const u16* Vbh = Vt + (size_t)bh * DP * SEQ;

  for (int kt = 0; kt < SEQ; kt += 32) {
    if (kt + 32 < SEQ) {                  // prefetch next K/V tiles
      __builtin_prefetch(Kbh + (size_t)(kt + 32 + nl) * DP, 0, 3);
      __builtin_prefetch(Vbh + (size_t)nl * SEQ + kt + 32, 0, 3);
    }
    // logits 16x32 as two 16x16 C tiles (K = depth = 64 -> 2 WMMA each)
    v8f c[2];
    #pragma unroll
    for (int j = 0; j < 2; ++j) {
      const u16* kp = Kbh + (size_t)(kt + 16 * j + nl) * DP + 8 * hf;
      v8f z = (v8f)(0.0f);
      z = wmma_bf16(qa0, ldB(kp), z);        // depth 0..31
      z = wmma_bf16(qa1, ldB(kp + 32), z);   // depth 32..63
      c[j] = z;
    }
    // online softmax in base-2; row (r+8*hf) spans this DPP row of 16 lanes
    #pragma unroll
    for (int r = 0; r < 8; ++r) {
      const float mx = row_max16(fmaxf(c[0][r], c[1][r]));
      const float mn = fmaxf(mrow[r], mx);
      const float al = __builtin_amdgcn_exp2f(mrow[r] - mn);
      mrow[r] = mn;
      const float p0 = __builtin_amdgcn_exp2f(c[0][r] - mn);
      const float p1 = __builtin_amdgcn_exp2f(c[1][r] - mn);
      c[0][r] = p0; c[1][r] = p1;
      lrow[r] = lrow[r] * al + row_sum16(p0 + p1);
      #pragma unroll
      for (int d = 0; d < 4; ++d) o[d][r] *= al;
    }
    // P (C layout, f32) -> LDS row-major [16][32] bf16 -> A fragment
    u16* W = &lds[wv][0];
    #pragma unroll
    for (int r = 0; r < 8; ++r) {
      const int rr = r + 8 * hf;
      W[rr * 32 + nl]      = f2bf(c[0][r]);
      W[rr * 32 + 16 + nl] = f2bf(c[1][r]);
    }
    // V fragment loads fly under the LDS round-trip (short live range)
    v16bf vb[4];
    #pragma unroll
    for (int d = 0; d < 4; ++d)
      vb[d] = ldB(Vbh + (size_t)(d * 16 + nl) * SEQ + kt + 8 * hf);
    asm volatile("s_wait_dscnt 0x0" ::: "memory");
    const v16bf pa = ldA(W + nl * 32 + 8 * hf);
    // O += P @ V
    #pragma unroll
    for (int d = 0; d < 4; ++d)
      o[d] = wmma_bf16(pa, vb[d], o[d]);
  }

  // normalize and store ctx as bf16 [B*S, D]
  #pragma unroll
  for (int r = 0; r < 8; ++r) {
    const float inv = 1.0f / lrow[r];
    const size_t row = (size_t)b * SEQ + qt + r + 8 * hf;
    #pragma unroll
    for (int d = 0; d < 4; ++d) {
      const int col = hh * DP + d * 16 + nl;
      ctx[row * DIMD + col] = f2bf(o[d][r] * inv);
    }
  }
}

// ---------- host ----------
extern "C" void kernel_launch(void* const* d_in, const int* in_sizes, int n_in,
                              void* d_out, int out_size, void* d_ws, size_t ws_size,
                              hipStream_t stream) {
  const float* v_in = (const float*)d_in[0];
  const float* k_in = (const float*)d_in[1];
  const float* q_in = (const float*)d_in[2];
  const float* Wq = (const float*)d_in[3];  const float* bq = (const float*)d_in[4];
  const float* Wk = (const float*)d_in[5];  const float* bk = (const float*)d_in[6];
  const float* Wv = (const float*)d_in[7];  const float* bv = (const float*)d_in[8];
  const float* Wo = (const float*)d_in[9];  const float* bo = (const float*)d_in[10];

  const size_t MB = 1ull << 20;
  if (ws_size < 40 * MB) return;
  char* ws = (char*)d_ws;
  u16* Wqt = (u16*)(ws + 0 * MB);
  u16* Wkt = (u16*)(ws + 2 * MB);
  u16* Wvt = (u16*)(ws + 4 * MB);
  u16* Wot = (u16*)(ws + 6 * MB);
  u16* Qh  = (u16*)(ws + 8 * MB);
  u16* Kh  = (u16*)(ws + 16 * MB);
  u16* Vt  = (u16*)(ws + 24 * MB);
  u16* ctx = (u16*)(ws + 32 * MB);

  dim3 bpack(256), gpack(DIMD * DIMD / 256);
  pack_wt<<<gpack, bpack, 0, stream>>>(Wq, Wqt);
  pack_wt<<<gpack, bpack, 0, stream>>>(Wk, Wkt);
  pack_wt<<<gpack, bpack, 0, stream>>>(Wv, Wvt);
  pack_wt<<<gpack, bpack, 0, stream>>>(Wo, Wot);

  // Q carries 1/sqrt(depth) * log2(e) so attention works in the exp2 domain
  const float qscale = 0.125f * 1.44269504088896f;

  dim3 gg(512), gb(32, 8);   // 512 blocks = 32 row-blocks x 16 col tiles
  gemm_kern<true ><<<gg, gb, 0, stream>>>(q_in, nullptr, Wqt, bq, Qh, nullptr,
                                          0, qscale);
  gemm_kern<true ><<<gg, gb, 0, stream>>>(k_in, nullptr, Wkt, bk, Kh, nullptr,
                                          0, 1.0f);
  gemm_kern<true ><<<gg, gb, 0, stream>>>(v_in, nullptr, Wvt, bv, Vt, nullptr,
                                          1, 1.0f);

  attn_kernel<<<dim3(NB * NH, SEQ / 64), dim3(32, 4), 0, stream>>>(Qh, Kh, Vt, ctx);

  gemm_kern<false><<<gg, gb, 0, stream>>>(nullptr, ctx, Wot, bo, nullptr,
                                          (float*)d_out, 2, 1.0f);
}
